// MultiHeadAttention_82592221102371
// MI455X (gfx1250) — matrix operand order repaired and checked by reading, not compiled, not run on hardware
//
#include <hip/hip_runtime.h>
#include <hip/hip_bf16.h>

// ---------------------------------------------------------------------------
// MHA forward for MI455X (gfx1250, wave32, WMMA + async-LDS + NT streaming).
//
// WMMA wave32 operand layouts (CDNA5 ISA 7.12.2):
//   A (16x32 bf16): lane L -> row m=L&15, half=L>>4;
//     elems 0..7:  K = 8*half + i        ; elems 8..15: K = 16 + 8*half + i-8
//   B (32x16 bf16): lane L -> col n=L&15; same K order as A
//   C/D (16x16 f32): lane L -> col n=L&15; VGPR j -> row m = j + 8*half.
// ---------------------------------------------------------------------------

typedef __attribute__((ext_vector_type(16))) __bf16        bf16x16;
typedef __attribute__((ext_vector_type(8)))  __bf16        bf16x8;
typedef __attribute__((ext_vector_type(8)))  float         f32x8;
typedef __attribute__((ext_vector_type(4)))  float         f32x4;
typedef __attribute__((ext_vector_type(4)))  unsigned short u16x4;

#define NB   8
#define NS   1024
#define ND   768
#define NH   12
#define NDEP 64
#define NM   (NB * NS)

#define KC   64            // K-chunk per LDS stage (2 WMMAs deep)
#define LP   72            // LDS row pitch in bf16 elems (144B, 16B aligned)

__device__ __forceinline__ unsigned short f2bf(float f) {
    __bf16 h = (__bf16)f;                       // native cvt, RNE
    return __builtin_bit_cast(unsigned short, h);
}
__device__ __forceinline__ u16x4 cvt4bf(f32x4 f) {
    u16x4 r;
    r.x = f2bf(f.x); r.y = f2bf(f.y); r.z = f2bf(f.z); r.w = f2bf(f.w);
    return r;
}
__device__ __forceinline__ f32x4 vmax4(f32x4 a, f32x4 b) {
    f32x4 r;
    r.x = fmaxf(a.x, b.x); r.y = fmaxf(a.y, b.y);
    r.z = fmaxf(a.z, b.z); r.w = fmaxf(a.w, b.w);
    return r;
}

__device__ __forceinline__ bf16x16 ld_a(const unsigned short* row, int half) {
    union { bf16x16 v; bf16x8 h[2]; } u;
    u.h[0] = *(const bf16x8*)(row + 8 * half);
    u.h[1] = *(const bf16x8*)(row + 8 * half + 16);
    return u.v;
}
__device__ __forceinline__ bf16x16 ld_b(const unsigned short* col, int half) {
    union { bf16x16 v; bf16x8 h[2]; } u;
    u.h[0] = *(const bf16x8*)(col + 8 * half);
    u.h[1] = *(const bf16x8*)(col + 8 * half + 16);
    return u.v;
}
__device__ __forceinline__ f32x8 wmma_bf16(bf16x16 a, bf16x16 b, f32x8 c) {
    return __builtin_amdgcn_wmma_f32_16x16x32_bf16(false, a, false, b,
                                                   (short)0, c, false, false);
}

// Shared inner step for the staged GEMMs: preload all B operands, then issue
// the 8 WMMAs back-to-back (single dscnt wait, WMMA pipeline stays fed).
__device__ __forceinline__ void gemm_step(
    const unsigned short* As, const unsigned short* Bs,
    int wave, int lane16, int half, f32x8 acc[4])
{
    const unsigned short* ar = &As[(wave * 16 + lane16) * LP];
    bf16x16 a0 = ld_a(ar, half);
    bf16x16 a1 = ld_a(ar + 32, half);
    bf16x16 b0[4], b1[4];
    #pragma unroll
    for (int j = 0; j < 4; ++j) {
        const unsigned short* bcol = &Bs[(j * 16 + lane16) * LP];
        b0[j] = ld_b(bcol, half);
        b1[j] = ld_b(bcol + 32, half);
    }
    #pragma unroll
    for (int j = 0; j < 4; ++j) {
        acc[j] = wmma_bf16(a0, b0[j], acc[j]);
        acc[j] = wmma_bf16(a1, b1[j], acc[j]);
    }
}

// ---------------------------------------------------------------------------
// Kernel 1: Y = X @ W + bias (8192x768x768) fp32 in -> bf16 head-split out.
// Tile 128x64, 256 thr (8 waves), K-chunk 64.
// ---------------------------------------------------------------------------
__global__ __launch_bounds__(256) void k_proj_gemm(
    const float* __restrict__ X, const float* __restrict__ W,
    const float* __restrict__ bias, unsigned short* __restrict__ Yh)
{
    __shared__ __align__(16) unsigned short As[128 * LP];
    __shared__ __align__(16) unsigned short Bs[64 * LP];

    const int t = threadIdx.x;
    const int wave = t >> 5, lane = t & 31;
    const int lane16 = lane & 15, half = lane >> 4;
    const int mbase = (int)blockIdx.x * 128;
    const int nbase = (int)blockIdx.y * 64;

    f32x8 acc[4] = {};

    for (int kc = 0; kc < ND; kc += KC) {
        // A: 128x64 fp32 -> bf16, float4-vectorized, coalesced
        #pragma unroll
        for (int i = 0; i < 8; ++i) {
            int g = t + i * 256;
            int r = g >> 4, p = g & 15;
            f32x4 f = *(const f32x4*)&X[(size_t)(mbase + r) * ND + kc + p * 4];
            *(u16x4*)&As[r * LP + p * 4] = cvt4bf(f);
        }
        // B transposed: Bt[n][k] = W[kc+k][nbase+n]; k-quads, n-coalesced loads
        #pragma unroll
        for (int i = 0; i < 4; ++i) {
            int g = t + i * 256;
            int n = g & 63, kq = g >> 6;
            f32x4 f;
            f.x = W[(size_t)(kc + kq * 4 + 0) * ND + nbase + n];
            f.y = W[(size_t)(kc + kq * 4 + 1) * ND + nbase + n];
            f.z = W[(size_t)(kc + kq * 4 + 2) * ND + nbase + n];
            f.w = W[(size_t)(kc + kq * 4 + 3) * ND + nbase + n];
            *(u16x4*)&Bs[n * LP + kq * 4] = cvt4bf(f);
        }
        __syncthreads();
        gemm_step(As, Bs, wave, lane16, half, acc);
        __syncthreads();
    }

    #pragma unroll
    for (int j = 0; j < 4; ++j) {
        int ng = nbase + j * 16 + lane16;
        float bv = bias[ng];
        int hh = ng >> 6, dd = ng & 63;
        #pragma unroll
        for (int r = 0; r < 8; ++r) {
            int m = mbase + wave * 16 + r + 8 * half;
            int bb = m >> 10, ss = m & 1023;
            Yh[(((size_t)bb * NH + hh) * NS + ss) * NDEP + dd] = f2bf(acc[j][r] + bv);
        }
    }
}

// ---------------------------------------------------------------------------
// Kernel 2: logits = (Qh Kh^T)/8 + mask*(-1e9) + adjoin ; softmax ; attn out.
// Block = (b,h) x 32 q-rows x 1024 keys, 128 thr (4 waves).
// Q/K WMMA operands loaded straight from global in lane layout; K-tile
// operands software-pipelined (prefetch tile kt+1 during WMMA of tile kt).
// ---------------------------------------------------------------------------
#define QROWS 32
#define LGP   1040

__global__ __launch_bounds__(128) void k_scores_softmax(
    const unsigned short* __restrict__ qh, const unsigned short* __restrict__ kh,
    const float* __restrict__ mask, const float* __restrict__ adjoin,
    float* __restrict__ attn)
{
    __shared__ __align__(16) float lg[QROWS * LGP];
    __shared__ float red[QROWS * 4];
    __shared__ float rowstat[QROWS];

    const int t = threadIdx.x;
    const int wave = t >> 5, lane = t & 31;
    const int lane16 = lane & 15, half = lane >> 4;

    const int blk  = blockIdx.x;
    const int bh   = blk >> 5;
    const int qblk = blk & 31;
    const int b    = bh / NH;
    const int qbase = qblk * QROWS;

    const int wr = (wave & 1) * 16;
    const int kb = (wave >> 1) * 512;

    const unsigned short* qrow =
        qh + ((size_t)bh * NS + qbase + wr + lane16) * NDEP;
    bf16x16 a0 = ld_a(qrow, half);
    bf16x16 a1 = ld_a(qrow + 32, half);

    const unsigned short* kbase_p = kh + ((size_t)bh * NS + kb + lane16) * NDEP;
    bf16x16 nb0 = ld_b(kbase_p, half);
    bf16x16 nb1 = ld_b(kbase_p + 32, half);

    for (int kt = 0; kt < 32; ++kt) {
        bf16x16 b0 = nb0, b1 = nb1;
        if (kt < 31) {                                  // prefetch next tile
            const unsigned short* kn = kbase_p + (size_t)(kt + 1) * 16 * NDEP;
            nb0 = ld_b(kn, half);
            nb1 = ld_b(kn + 32, half);
        }
        f32x8 acc = {};
        acc = wmma_bf16(a0, b0, acc);
        acc = wmma_bf16(a1, b1, acc);
        int kr = kb + kt * 16;
        #pragma unroll
        for (int r = 0; r < 8; ++r)
            lg[(wr + r + 8 * half) * LGP + kr + lane16] = acc[r];
    }
    __syncthreads();

    // scale + mask + adjoin + row max (4 threads/row, 64 float4 each)
    const int row = t >> 2, q4 = t & 3, c0 = q4 * 256;
    const int qglob = qbase + row;
    const f32x4* m4 = (const f32x4*)&mask[(size_t)b * NS + c0];
    const f32x4* a4 = (const f32x4*)&adjoin[((size_t)b * NS + qglob) * NS + c0];
    f32x4* l4 = (f32x4*)&lg[row * LGP + c0];
    f32x4 mx4 = {-3.0e38f, -3.0e38f, -3.0e38f, -3.0e38f};
    for (int c = 0; c < 64; ++c) {
        f32x4 v = l4[c] * 0.125f + m4[c] * (-1e9f) + a4[c];
        l4[c] = v;
        mx4 = vmax4(mx4, v);
    }
    red[row * 4 + q4] = fmaxf(fmaxf(mx4.x, mx4.y), fmaxf(mx4.z, mx4.w));
    __syncthreads();
    if (q4 == 0)
        rowstat[row] = fmaxf(fmaxf(red[row * 4 + 0], red[row * 4 + 1]),
                             fmaxf(red[row * 4 + 2], red[row * 4 + 3]));
    __syncthreads();

    float rmax = rowstat[row];
    f32x4 sm4 = {0.f, 0.f, 0.f, 0.f};
    for (int c = 0; c < 64; ++c) {
        f32x4 v = l4[c];
        v.x = __expf(v.x - rmax); v.y = __expf(v.y - rmax);
        v.z = __expf(v.z - rmax); v.w = __expf(v.w - rmax);
        l4[c] = v;
        sm4 += v;
    }
    __syncthreads();
    red[row * 4 + q4] = sm4.x + sm4.y + sm4.z + sm4.w;
    __syncthreads();
    if (q4 == 0)
        rowstat[row] = red[row * 4 + 0] + red[row * 4 + 1]
                     + red[row * 4 + 2] + red[row * 4 + 3];
    __syncthreads();

    // normalized, coalesced, NON-TEMPORAL attn write (402MB stream > L2)
    float* aout = attn + ((size_t)bh * NS + qbase) * NS;
    for (int idx = t; idx < QROWS * (NS / 4); idx += 128) {
        int r = idx >> 8, c4 = idx & 255;
        f32x4 v = *(const f32x4*)&lg[r * LGP + c4 * 4] * (1.0f / rowstat[r]);
        __builtin_nontemporal_store(v, (f32x4*)&aout[(size_t)r * NS + c4 * 4]);
    }
}

// ---------------------------------------------------------------------------
// Kernel 3: ctx = attn @ Vh per (b,h): 1024x64x1024. Tile 128x64, 256 thr.
// attn read non-temporally (read-once stream), fp32 -> bf16 staged.
// ---------------------------------------------------------------------------
__global__ __launch_bounds__(256) void k_attn_v(
    const float* __restrict__ attn, const unsigned short* __restrict__ vh,
    unsigned short* __restrict__ ctx)
{
    __shared__ __align__(16) unsigned short As[128 * LP];
    __shared__ __align__(16) unsigned short Bs[64 * LP];

    const int t = threadIdx.x;
    const int wave = t >> 5, lane = t & 31;
    const int lane16 = lane & 15, half = lane >> 4;

    const int blk = blockIdx.x;
    const int bh  = blk >> 3;
    const int mt  = blk & 7;
    const int b = bh / NH, h = bh % NH;
    const int mbase = mt * 128;

    const float*          arow = attn + (size_t)bh * NS * NS;
    const unsigned short* vrow = vh   + (size_t)bh * NS * NDEP;

    f32x8 acc[4] = {};

    for (int kc = 0; kc < NS; kc += KC) {
        #pragma unroll
        for (int i = 0; i < 8; ++i) {
            int g = t + i * 256;
            int r = g >> 4, p = g & 15;
            f32x4 f = __builtin_nontemporal_load(
                (const f32x4*)&arow[(size_t)(mbase + r) * NS + kc + p * 4]);
            *(u16x4*)&As[r * LP + p * 4] = cvt4bf(f);
        }
        #pragma unroll
        for (int i = 0; i < 4; ++i) {
            int g = t + i * 256;
            int n = g & 63, kq = g >> 6;
            u16x4 u;
            u.x = vrow[(size_t)(kc + kq * 4 + 0) * NDEP + n];
            u.y = vrow[(size_t)(kc + kq * 4 + 1) * NDEP + n];
            u.z = vrow[(size_t)(kc + kq * 4 + 2) * NDEP + n];
            u.w = vrow[(size_t)(kc + kq * 4 + 3) * NDEP + n];
            *(u16x4*)&Bs[n * LP + kq * 4] = u;
        }
        __syncthreads();
        gemm_step(As, Bs, wave, lane16, half, acc);
        __syncthreads();
    }

    #pragma unroll
    for (int j = 0; j < 4; ++j) {
        int nl = j * 16 + lane16;
        #pragma unroll
        for (int r = 0; r < 8; ++r) {
            int s = mbase + wave * 16 + r + 8 * half;
            ctx[((size_t)b * NS + s) * ND + h * NDEP + nl] = f2bf(acc[j][r]);
        }
    }
}

// ---------------------------------------------------------------------------
// Kernel 4: output = ctx @ wo + bo (8192x768x768), bf16 A copied into LDS with
// CDNA5 async global->LDS DMA (ASYNCcnt path), fp32 out.
// ---------------------------------------------------------------------------
typedef __attribute__((address_space(3))) unsigned short lds_ushort;

__global__ __launch_bounds__(256) void k_out_gemm(
    const unsigned short* __restrict__ Xbf, const float* __restrict__ W,
    const float* __restrict__ bias, float* __restrict__ Y)
{
    __shared__ __align__(16) unsigned short As[128 * LP];
    __shared__ __align__(16) unsigned short Bs[64 * LP];

    const int t = threadIdx.x;
    const int wave = t >> 5, lane = t & 31;
    const int lane16 = lane & 15, half = lane >> 4;
    const int mbase = (int)blockIdx.x * 128;
    const int nbase = (int)blockIdx.y * 64;

    const unsigned long long gbase = (unsigned long long)Xbf;

    f32x8 acc[4] = {};

    for (int kc = 0; kc < ND; kc += KC) {
        // A: straight bf16 copy 128x64 -> async DMA to LDS, 8B per lane-op
        #pragma unroll
        for (int i = 0; i < 8; ++i) {
            int g = t + i * 256;
            int r = g >> 4, p = g & 15;
            unsigned goff = (unsigned)(((size_t)(mbase + r) * ND + kc + p * 4) * 2);
            lds_ushort* ldst = (lds_ushort*)As + (r * LP + p * 4);
            asm volatile("global_load_async_to_lds_b64 %0, %1, %2"
                         :: "v"(ldst), "v"(goff), "s"(gbase) : "memory");
        }
        // B transposed fp32 -> bf16 (regular path)
        #pragma unroll
        for (int i = 0; i < 4; ++i) {
            int g = t + i * 256;
            int n = g & 63, kq = g >> 6;
            f32x4 f;
            f.x = W[(size_t)(kc + kq * 4 + 0) * ND + nbase + n];
            f.y = W[(size_t)(kc + kq * 4 + 1) * ND + nbase + n];
            f.z = W[(size_t)(kc + kq * 4 + 2) * ND + nbase + n];
            f.w = W[(size_t)(kc + kq * 4 + 3) * ND + nbase + n];
            *(u16x4*)&Bs[n * LP + kq * 4] = cvt4bf(f);
        }
        asm volatile("s_wait_asynccnt 0" ::: "memory");
        __syncthreads();
        gemm_step(As, Bs, wave, lane16, half, acc);
        __syncthreads();
    }

    #pragma unroll
    for (int j = 0; j < 4; ++j) {
        int ng = nbase + j * 16 + lane16;
        float bv = bias[ng];
        #pragma unroll
        for (int r = 0; r < 8; ++r) {
            int m = mbase + wave * 16 + r + 8 * half;
            Y[(size_t)m * ND + ng] = acc[j][r] + bv;
        }
    }
}

// ---------------------------------------------------------------------------
extern "C" void kernel_launch(void* const* d_in, const int* in_sizes, int n_in,
                              void* d_out, int out_size, void* d_ws, size_t ws_size,
                              hipStream_t stream) {
    const float* v      = (const float*)d_in[0];
    const float* k      = (const float*)d_in[1];
    const float* q      = (const float*)d_in[2];
    const float* mask   = (const float*)d_in[3];
    const float* adjoin = (const float*)d_in[4];
    const float* wq = (const float*)d_in[5];  const float* bq = (const float*)d_in[6];
    const float* wk = (const float*)d_in[7];  const float* bk = (const float*)d_in[8];
    const float* wv = (const float*)d_in[9];  const float* bv = (const float*)d_in[10];
    const float* wo = (const float*)d_in[11]; const float* bo = (const float*)d_in[12];

    float* out  = (float*)d_out;                    // (B,S,D)
    float* attn = out + (size_t)NM * ND;            // (B,H,S,S)

    const size_t HE = (size_t)NB * NH * NS * NDEP;
    unsigned short* qh  = (unsigned short*)d_ws;
    unsigned short* kh  = qh + HE;
    unsigned short* vh  = kh + HE;
    unsigned short* ctx = vh + HE;

    dim3 gproj(NM / 128, ND / 64);
    k_proj_gemm<<<gproj, 256, 0, stream>>>(q, wq, bq, qh);
    k_proj_gemm<<<gproj, 256, 0, stream>>>(k, wk, bk, kh);
    k_proj_gemm<<<gproj, 256, 0, stream>>>(v, wv, bv, vh);

    k_scores_softmax<<<NB * NH * (NS / QROWS), 128, 0, stream>>>(
        qh, kh, mask, adjoin, attn);

    k_attn_v<<<NB * NH * (NS / 128), 256, 0, stream>>>(attn, vh, ctx);

    k_out_gemm<<<gproj, 256, 0, stream>>>(ctx, wo, bo, out);
}
